// LinearAttention_37563783970830
// MI455X (gfx1250) — matrix operand order repaired and checked by reading, not compiled, not run on hardware
//
#include <hip/hip_runtime.h>
#include <hip/hip_bf16.h>
#include <stdint.h>

// ---------------------------------------------------------------------------
// LinearAttention on MI455X (gfx1250): bf16 WMMA GEMMs, TDM-staged A tiles.
// HBM-bound (~200MB traffic @ 23.3TB/s ~ 9us); GEMMs (~52 GFLOP) trivial for
// the WMMA pipes. A-tiles (activations) are DMA'd to LDS once per workgroup
// via the Tensor Data Mover and shared by 8 waves; B (weights) streams from
// L2 (1.5MB bf16 total).
// ---------------------------------------------------------------------------

#define DIM       512
#define SEQ       8192
#define BATCH     4
#define BS        (BATCH * SEQ)       // 32768 token rows
#define NH        8
#define NKV       4
#define HD        64                  // head dim
#define QK_SCALE  0.125f              // 1/sqrt(64)

typedef __attribute__((ext_vector_type(16))) __bf16       v16bf;
typedef __attribute__((ext_vector_type(8)))  float        v8f;
typedef __attribute__((ext_vector_type(4)))  unsigned int u32x4;
typedef __attribute__((ext_vector_type(4)))  int          i32x4;
typedef __attribute__((ext_vector_type(8)))  int          i32x8;

union FragAB {              // 16 bf16 = 32 bytes = 8 VGPRs per lane
    v16bf bf;
    u32x4 u[2];
};

#if defined(__has_builtin)
#  if __has_builtin(__builtin_amdgcn_tensor_load_to_lds) && \
      __has_builtin(__builtin_amdgcn_s_wait_tensorcnt)
#    define USE_TDM 1
#  endif
#endif
#ifndef USE_TDM
#  define USE_TDM 0
#endif

__device__ __forceinline__ unsigned short f2bf(float f) {
    unsigned int u = __float_as_uint(f);
    u += 0x7FFFu + ((u >> 16) & 1u);        // round-to-nearest-even
    return (unsigned short)(u >> 16);
}
__device__ __forceinline__ float bf2f(unsigned short h) {
    return __uint_as_float(((unsigned int)h) << 16);
}
// phi(x) = max(elu(x), -0.99) + 1  ==  x>0 ? x+1 : max(exp(x), 0.01)
__device__ __forceinline__ float phi(float x) {
    return x > 0.f ? x + 1.f : fmaxf(__expf(x), 0.01f);
}

#if USE_TDM
// Tensor DMA: load one 16-row x 1024-byte tile (16 x 512 bf16) into LDS.
// D# group0: count=1 | lds_addr | global_addr[56:0] | type=2.
// D# group1: data_size=8B units, tensor row = 128 units, tile = 128x16,
//            row stride = 128 units, workgroup_mask=0 (not in a cluster).
__device__ __forceinline__ void tdm_load_16x1024B(unsigned lds_off,
                                                  const void* gsrc) {
    unsigned long long ga = (unsigned long long)(uintptr_t)gsrc;
    u32x4 g0;
    g0[0] = 1u;                                       // count=1, user mode
    g0[1] = lds_off;                                  // LDS byte address
    g0[2] = (unsigned)(ga & 0xFFFFFFFFu);             // global_addr[31:0]
    g0[3] = (unsigned)((ga >> 32) & 0x01FFFFFFu)      // global_addr[56:32]
          | 0x80000000u;                              // type=2 ("image")
    i32x8 g1;
    g1[0] = (int)(3u << 16);                          // data_size = 8 bytes
    g1[1] = (int)(128u << 16);                        // tensor_dim0 = 128 units
    g1[2] = (int)(32768u << 16);                      // tensor_dim1 = BS rows (lo16)
    g1[3] = (int)(128u << 16);                        // tile_dim0 = 128 units
    g1[4] = 16;                                       // tile_dim1 = 16 rows
    g1[5] = 128;                                      // tensor_dim0_stride = 128
    g1[6] = 0;
    g1[7] = 0;
    i32x4 z4 = {0, 0, 0, 0};
#if defined(__clang_major__) && (__clang_major__ >= 23)
    i32x8 z8 = {0, 0, 0, 0, 0, 0, 0, 0};
    __builtin_amdgcn_tensor_load_to_lds(g0, g1, z4, z4, z8, 0);
#else
    __builtin_amdgcn_tensor_load_to_lds(g0, g1, z4, z4, 0);
#endif
}
#endif

// Stage a 16x512 bf16 tile (16KB) into LDS, visible to all waves afterwards.
__device__ __forceinline__ void stage_rows16(unsigned short* As /*LDS*/,
                                             const unsigned short* src) {
#if USE_TDM
    if (threadIdx.x < 32) {                 // wave 0 issues the DMA and waits
        tdm_load_16x1024B((unsigned)(uintptr_t)As, src);
        __builtin_amdgcn_s_wait_tensorcnt(0);
    }
    __syncthreads();
#else
    const u32x4* s = (const u32x4*)src;
    u32x4* d = (u32x4*)As;
    for (int i = threadIdx.x; i < (16 * DIM) / 8; i += 256) d[i] = s[i];
    __syncthreads();
#endif
}

// A-fragment (16x32 bf16) from the LDS-resident tile (row-major, ld=512).
// lane L: M = L%16, kb = (L/16)*8; elems 0..7 = K[kb..+7], 8..15 = K[kb+16..+23]
__device__ __forceinline__ void load_fragA_lds(FragAB& a,
                                               const unsigned short* As,
                                               int k0) {
    int lane = threadIdx.x & 31;
    const unsigned short* p = As + (lane & 15) * DIM + k0 + ((lane >> 4) << 3);
    a.u[0] = *(const u32x4*)(p);
    a.u[1] = *(const u32x4*)(p + 16);
}
// B-fragment: 32x16 bf16 from W[n][d] row-major (ld=512): B[k][n] = W[n][k].
__device__ __forceinline__ void load_fragB(FragAB& b, const unsigned short* W,
                                           int n0, int k0) {
    int lane = threadIdx.x & 31;
    const unsigned short* p = W + (size_t)(n0 + (lane & 15)) * DIM
                                + k0 + ((lane >> 4) << 3);
    b.u[0] = *(const u32x4*)(p);
    b.u[1] = *(const u32x4*)(p + 16);
}

__device__ __forceinline__ v8f wmma_bf16(const FragAB& a, const FragAB& b, v8f c) {
    return __builtin_amdgcn_wmma_f32_16x16x32_bf16(
        /*neg_a=*/false, a.bf, /*neg_b=*/false, b.bf,
        /*c_mod=*/(short)0, c, /*reuse_a=*/false, /*reuse_b=*/false);
}

// ---------------------------------------------------------------------------
// K0a: weight prep -> bf16.  Wc[1024][512] = [wq; wk; wv], Wo[512][512] = wo.
// ---------------------------------------------------------------------------
__global__ __launch_bounds__(256) void wprep_kernel(
        const float* __restrict__ wq, const float* __restrict__ wk,
        const float* __restrict__ wv, const float* __restrict__ wo,
        unsigned short* __restrict__ Wc, unsigned short* __restrict__ Wo) {
    int i = blockIdx.x * 256 + threadIdx.x;
    if (i < 1024 * DIM) {
        int n = i >> 9, d = i & 511;
        float v;
        if (n < 512)      v = wq[(size_t)n * DIM + d];
        else if (n < 768) v = wk[(size_t)(n - 512) * DIM + d];
        else              v = wv[(size_t)(n - 768) * DIM + d];
        Wc[i] = f2bf(v);
    }
    if (i < 512 * DIM) Wo[i] = f2bf(wo[i]);
}

// K0b: zero the KV / K_sum accumulators (2*B*NKV*HD = 2048 floats).
__global__ __launch_bounds__(256) void zero_kernel(float* __restrict__ acc) {
    int i = blockIdx.x * 256 + threadIdx.x;
    if (i < 2 * BATCH * NKV * HD) acc[i] = 0.f;
}

// ---------------------------------------------------------------------------
// K1: RMSNorm -> h (bf16).  One wave32 per row; lane owns 16 strided elems.
// ---------------------------------------------------------------------------
__global__ __launch_bounds__(256) void rmsnorm_kernel(
        const float* __restrict__ x, const float* __restrict__ g,
        unsigned short* __restrict__ h) {
    int wave = threadIdx.x >> 5, lane = threadIdx.x & 31;
    size_t row = (size_t)blockIdx.x * 8 + wave;
    const float* xr = x + row * DIM;
    float vals[16];
    float ss = 0.f;
#pragma unroll
    for (int i = 0; i < 16; ++i) {
        float v = xr[lane + i * 32];
        vals[i] = v;
        ss += v * v;
    }
#pragma unroll
    for (int o = 16; o >= 1; o >>= 1) ss += __shfl_xor(ss, o, 32);
    float rinv = rsqrtf(ss * (1.f / DIM) + 1e-6f);
    unsigned short* hr = h + row * DIM;
#pragma unroll
    for (int i = 0; i < 16; ++i) {
        int d = lane + i * 32;
        hr[d] = f2bf(vals[i] * rinv * g[d]);
    }
}

// ---------------------------------------------------------------------------
// K2: QKV GEMM via WMMA bf16.  grid = BS/16, block = 256 (8 waves).
// A tile (16x512 bf16) is TDM-staged in LDS once and shared by all waves.
//   waves 0-3: 8 Q n-tiles each  -> scale, phi, store Q bf16
//   waves 4-7: 4 paired K/V col-tiles each -> phi(K)*V and phi(K) reduced
//              over the 16 s-rows, atomicAdd into KV / Ksum (2048 floats).
// ---------------------------------------------------------------------------
__global__ __launch_bounds__(256) void qkv_kernel(
        const unsigned short* __restrict__ h, const unsigned short* __restrict__ Wc,
        unsigned short* __restrict__ Q, float* __restrict__ KVacc,
        float* __restrict__ KSacc) {
    __shared__ unsigned short As[16 * DIM];
    const int m0   = blockIdx.x * 16;
    const int wave = threadIdx.x >> 5;
    const int lane = threadIdx.x & 31;

    stage_rows16(As, h + (size_t)m0 * DIM);

    if (wave < 4) {                              // ---- Q path: tiles wave*8+i
        v8f acc[8] = {};
        for (int k0 = 0; k0 < DIM; k0 += 32) {
            FragAB a;
            load_fragA_lds(a, As, k0);
#pragma unroll
            for (int i = 0; i < 8; ++i) {
                FragAB b;
                load_fragB(b, Wc, (wave * 8 + i) * 16, k0);
                acc[i] = wmma_bf16(a, b, acc[i]);
            }
        }
        int mrow = m0 + ((lane >> 4) << 3);
#pragma unroll
        for (int i = 0; i < 8; ++i) {
            int n = (wave * 8 + i) * 16 + (lane & 15);
#pragma unroll
            for (int r = 0; r < 8; ++r) {
                float q = acc[i][r] * QK_SCALE;
                Q[(size_t)(mrow + r) * DIM + n] = f2bf(phi(q));
            }
        }
    } else {                                     // ---- K/V path: tiles (wave-4)*4+i
        const int tb = (wave - 4) * 4;
        v8f ak[4] = {}, av[4] = {};
        for (int k0 = 0; k0 < DIM; k0 += 32) {
            FragAB a;
            load_fragA_lds(a, As, k0);
#pragma unroll
            for (int i = 0; i < 4; ++i) {
                FragAB bk, bv;
                load_fragB(bk, Wc, 512 + (tb + i) * 16, k0);
                load_fragB(bv, Wc, 768 + (tb + i) * 16, k0);
                ak[i] = wmma_bf16(a, bk, ak[i]);
                av[i] = wmma_bf16(a, bv, av[i]);
            }
        }
        const int b = m0 / SEQ;                  // whole 16-row tile is same batch
#pragma unroll
        for (int i = 0; i < 4; ++i) {
            int c = (tb + i) * 16 + (lane & 15); // kv feature col 0..255
            int g = c >> 6, d = c & 63;
            float psum = 0.f, ksum = 0.f;
#pragma unroll
            for (int r = 0; r < 8; ++r) {
                float kq = phi(ak[i][r] * QK_SCALE);
                psum += kq * av[i][r];
                ksum += kq;
            }
            psum += __shfl_xor(psum, 16, 32);    // fold the two M-halves (same N)
            ksum += __shfl_xor(ksum, 16, 32);
            if (lane < 16) {
                int idx = (b * NKV + g) * HD + d;
                atomicAdd(&KVacc[idx], psum);
                atomicAdd(&KSacc[idx], ksum);
            }
        }
    }
}

// ---------------------------------------------------------------------------
// K3: Y = phi(Q) * KV / (Q . Ksum + eps)  per (row, head).  One wave per head.
// ---------------------------------------------------------------------------
__global__ __launch_bounds__(256) void y_kernel(
        const unsigned short* __restrict__ Q, const float* __restrict__ KVacc,
        const float* __restrict__ KSacc, unsigned short* __restrict__ Y) {
    int row  = blockIdx.x;
    int hh   = threadIdx.x >> 5;
    int lane = threadIdx.x & 31;
    int b    = row / SEQ;
    int g    = hh >> 1;                           // GQA: kv group of this head
    size_t base = ((size_t)row * DIM + hh * HD) >> 1;  // in uint units
    unsigned int qp = ((const unsigned int*)Q)[base + lane];
    float q0 = bf2f((unsigned short)(qp & 0xFFFFu));
    float q1 = bf2f((unsigned short)(qp >> 16));
    int d0 = 2 * lane, d1 = 2 * lane + 1;
    const float* ks = KSacc + (b * NKV + g) * HD;
    const float* kv = KVacc + (b * NKV + g) * HD;
    float z = q0 * ks[d0] + q1 * ks[d1];
#pragma unroll
    for (int o = 16; o >= 1; o >>= 1) z += __shfl_xor(z, o, 32);
    float inv = 1.f / (z + 1e-6f);
    float y0 = q0 * kv[d0] * inv;
    float y1 = q1 * kv[d1] * inv;
    ((unsigned int*)Y)[base + lane] =
        ((unsigned int)f2bf(y1) << 16) | (unsigned int)f2bf(y0);
}

// ---------------------------------------------------------------------------
// K4: out = Y @ wo^T via WMMA bf16, fp32 out.  grid = BS/16, block = 256.
// A tile TDM-staged in LDS; 8 waves x 4 n-tiles cover all 512 out cols.
// ---------------------------------------------------------------------------
__global__ __launch_bounds__(256) void outgemm_kernel(
        const unsigned short* __restrict__ Y, const unsigned short* __restrict__ Wo,
        float* __restrict__ out) {
    __shared__ unsigned short As[16 * DIM];
    const int m0   = blockIdx.x * 16;
    const int wave = threadIdx.x >> 5;
    const int lane = threadIdx.x & 31;

    stage_rows16(As, Y + (size_t)m0 * DIM);

    v8f acc[4] = {};
    for (int k0 = 0; k0 < DIM; k0 += 32) {
        FragAB a;
        load_fragA_lds(a, As, k0);
#pragma unroll
        for (int i = 0; i < 4; ++i) {
            FragAB b;
            load_fragB(b, Wo, (wave * 4 + i) * 16, k0);
            acc[i] = wmma_bf16(a, b, acc[i]);
        }
    }
    int mrow = m0 + ((lane >> 4) << 3);
#pragma unroll
    for (int i = 0; i < 4; ++i) {
        int n = (wave * 4 + i) * 16 + (lane & 15);
#pragma unroll
        for (int r = 0; r < 8; ++r)
            out[(size_t)(mrow + r) * DIM + n] = acc[i][r];
    }
}

// ---------------------------------------------------------------------------
// Workspace layout (bytes): Wc bf16 | Wo bf16 | KV+Ksum f32 | h/Y bf16 | Q bf16
// ---------------------------------------------------------------------------
static const size_t OFF_WC = 0;                                   // 1 MB
static const size_t OFF_WO = 1u << 20;                            // 512 KB
static const size_t OFF_KV = OFF_WO + (size_t)512 * 512 * 2;      // 4 KB
static const size_t OFF_KS = OFF_KV + 4096;                       // 4 KB
static const size_t OFF_H  = 2u << 20;                            // 32 MB (h, reused as Y)
static const size_t OFF_Q  = OFF_H + (size_t)BS * DIM * 2;        // 32 MB

extern "C" void kernel_launch(void* const* d_in, const int* in_sizes, int n_in,
                              void* d_out, int out_size, void* d_ws, size_t ws_size,
                              hipStream_t stream) {
    (void)in_sizes; (void)n_in; (void)out_size; (void)ws_size;
    const float* x  = (const float*)d_in[0];
    const float* g  = (const float*)d_in[1];
    const float* wq = (const float*)d_in[2];
    const float* wk = (const float*)d_in[3];
    const float* wv = (const float*)d_in[4];
    const float* wo = (const float*)d_in[5];
    float* out = (float*)d_out;

    uint8_t* ws = (uint8_t*)d_ws;
    unsigned short* Wc = (unsigned short*)(ws + OFF_WC);
    unsigned short* Wo = (unsigned short*)(ws + OFF_WO);
    float*          KV = (float*)(ws + OFF_KV);
    float*          KS = (float*)(ws + OFF_KS);
    unsigned short* H  = (unsigned short*)(ws + OFF_H);   // h, then reused as Y
    unsigned short* Q  = (unsigned short*)(ws + OFF_Q);

    wprep_kernel<<<dim3((1024 * DIM + 255) / 256), dim3(256), 0, stream>>>(
        wq, wk, wv, wo, Wc, Wo);
    zero_kernel<<<dim3(8), dim3(256), 0, stream>>>(KV);   // KV and KS contiguous
    rmsnorm_kernel<<<dim3(BS / 8), dim3(256), 0, stream>>>(x, g, H);
    qkv_kernel<<<dim3(BS / 16), dim3(256), 0, stream>>>(H, Wc, Q, KV, KS);
    y_kernel<<<dim3(BS), dim3(256), 0, stream>>>(Q, KV, KS, H /* Y reuses h */);
    outgemm_kernel<<<dim3(BS / 16), dim3(256), 0, stream>>>(H, Wo, out);
}
